// MultiHeadAttention_3599182594711
// MI455X (gfx1250) — matrix operand order repaired and checked by reading, not compiled, not run on hardware
//
#include <hip/hip_runtime.h>
#include <hip/hip_bf16.h>

typedef __attribute__((ext_vector_type(16))) _Float16 v16h;
typedef __attribute__((ext_vector_type(8)))  _Float16 v8h;
typedef __attribute__((ext_vector_type(8)))  float    v8f;
typedef __attribute__((ext_vector_type(4)))  unsigned int v4u;
typedef __attribute__((ext_vector_type(8)))  int      v8i;
typedef __attribute__((ext_vector_type(4)))  int      v4i;

#define BATCH 2
#define SEQ   2048
#define DMODEL 1024
#define NHEAD 16
#define DKH   64
#define MROWS (BATCH*SEQ)   // 4096
#define KCHUNK 64           // keys staged per TDM tile

__device__ __forceinline__ v8f wmma_f16(v16h a, v16h b, v8f c) {
    return __builtin_amdgcn_wmma_f32_16x16x32_f16(false, a, false, b, (short)0, c, false, false);
}

// ---------------------------------------------------------------------------
// Tensor Data Mover: 2D tile (f16, data_size code 1) global -> LDS.
// Descriptor bitfields per CDNA5 ISA.
// ---------------------------------------------------------------------------
__device__ __forceinline__ void tdm_load_2d_f16(unsigned lds_addr, const void* gptr,
                                                unsigned tile_d0, unsigned tile_d1,
                                                unsigned tens_d0, unsigned tens_d1,
                                                unsigned long long stride0)
{
    unsigned long long ga = (unsigned long long)(uintptr_t)gptr;
    v4u g0;
    g0[0] = 1u;                                   // count=1, user descriptor
    g0[1] = lds_addr;                             // LDS byte address
    g0[2] = (unsigned)(ga & 0xFFFFFFFFu);         // global_addr[31:0]
    g0[3] = (unsigned)((ga >> 32) & 0x1FFFFFFu)   // global_addr[56:32]
          | (2u << 30);                           // type = 2 ("image")
    v8i g1;
    g1[0] = (int)(1u << 16);                      // data_size = 1 (2 bytes)
    g1[1] = (int)((tens_d0 & 0xFFFFu) << 16);                                  // tensor_dim0[15:0]
    g1[2] = (int)(((tens_d0 >> 16) & 0xFFFFu) | ((tens_d1 & 0xFFFFu) << 16));  // dim0 hi | dim1 lo
    g1[3] = (int)(((tens_d1 >> 16) & 0xFFFFu) | ((tile_d0 & 0xFFFFu) << 16));  // dim1 hi | tile_dim0
    g1[4] = (int)(tile_d1 & 0xFFFFu);                                          // tile_dim1, tile_dim2=0
    g1[5] = (int)(unsigned)(stride0 & 0xFFFFFFFFu);                            // tensor_dim0_stride lo
    g1[6] = (int)(unsigned)((stride0 >> 32) & 0xFFFFu);                        // stride hi
    g1[7] = 0;
    v4i z4 = {0, 0, 0, 0};
    v8i z8 = {0, 0, 0, 0, 0, 0, 0, 0};
    __builtin_amdgcn_tensor_load_to_lds(g0, g1, z4, z4, z8, 0);
}

__device__ __forceinline__ unsigned lds_byte_offset(const void* p) {
    return (unsigned)(uintptr_t)p;   // low 32 bits of flat addr = LDS byte offset
}

// ---------------------------------------------------------------------------
// One-shot f32 -> f16 conversion (inputs + weights), 8 elements/thread.
// ---------------------------------------------------------------------------
__global__ __launch_bounds__(256) void cvt_kernel(const float* __restrict__ in,
                                                  _Float16* __restrict__ out, int n8)
{
    const int i = blockIdx.x * blockDim.x + threadIdx.x;
    if (i >= n8) return;
    const float4* p = (const float4*)in + (size_t)i * 2;
    const float4 a = p[0], b = p[1];
    v8h o;
    o[0] = (_Float16)a.x; o[1] = (_Float16)a.y; o[2] = (_Float16)a.z; o[3] = (_Float16)a.w;
    o[4] = (_Float16)b.x; o[5] = (_Float16)b.y; o[6] = (_Float16)b.z; o[7] = (_Float16)b.w;
    *(v8h*)(out + (size_t)i * 8) = o;
}

// A-tile loader (f16 row-major source): 16x32, lane = g*16+row
__device__ __forceinline__ v16h load_a16(const _Float16* row, int kk, int g) {
    v8h lo = *(const v8h*)(row + kk + 16 * g);
    v8h hi = *(const v8h*)(row + kk + 16 * g + 8);
    v16h a;
    #pragma unroll
    for (int i = 0; i < 8; ++i) { a[i] = lo[i]; a[i + 8] = hi[i]; }
    return a;
}

// ---------------------------------------------------------------------------
// Projection GEMM (all-f16 operands): out = (X @ W^T + bias) * scale, f16.
// Wave tile 32(M) x 64(N); B tile shared by both M sub-tiles.
// vtrans=0 -> [B,H,L,DK]; vtrans=1 -> [B,H,DK,L] (used for V).
// grid = (128, 2), block = 256.
// ---------------------------------------------------------------------------
__global__ __launch_bounds__(256) void proj_kernel(
    const _Float16* __restrict__ X, const _Float16* __restrict__ W,
    const float* __restrict__ bias, _Float16* __restrict__ out,
    float scale, int vtrans)
{
    const int lane = threadIdx.x & 31;
    const int wave = threadIdx.x >> 5;
    const int g  = lane >> 4;
    const int ln = lane & 15;
    const int m0 = blockIdx.x * 32;
    const int n0 = blockIdx.y * 512 + wave * 64;

    const _Float16* xrow0 = X + (size_t)(m0 + ln) * DMODEL;
    const _Float16* xrow1 = X + (size_t)(m0 + 16 + ln) * DMODEL;

    v8f aA0 = {}, aA1 = {}, aA2 = {}, aA3 = {};   // rows m0..m0+15
    v8f aB0 = {}, aB1 = {}, aB2 = {}, aB3 = {};   // rows m0+16..m0+31

    for (int kk = 0; kk < DMODEL; kk += 32) {
        __builtin_prefetch(xrow0 + kk + 256, 0, 0);   // -> global_prefetch_b8
        const v16h a0 = load_a16(xrow0, kk, g);
        const v16h a1 = load_a16(xrow1, kk, g);
        #pragma unroll
        for (int nt = 0; nt < 4; ++nt) {
            const v16h bm = *(const v16h*)(W + (size_t)(n0 + nt * 16 + ln) * DMODEL + kk + 16 * g);
            v8f* pA = (nt == 0) ? &aA0 : (nt == 1) ? &aA1 : (nt == 2) ? &aA2 : &aA3;
            v8f* pB = (nt == 0) ? &aB0 : (nt == 1) ? &aB1 : (nt == 2) ? &aB2 : &aB3;
            *pA = wmma_f16(a0, bm, *pA);
            *pB = wmma_f16(a1, bm, *pB);
        }
    }

    #pragma unroll
    for (int nt = 0; nt < 4; ++nt) {
        const int n  = n0 + nt * 16 + ln;
        const float bb = bias[n];
        const int hh = n >> 6, dk = n & 63;
        const v8f accA = (nt == 0) ? aA0 : (nt == 1) ? aA1 : (nt == 2) ? aA2 : aA3;
        const v8f accB = (nt == 0) ? aB0 : (nt == 1) ? aB1 : (nt == 2) ? aB2 : aB3;
        #pragma unroll
        for (int sub = 0; sub < 2; ++sub) {
            const v8f acc = sub ? accB : accA;
            #pragma unroll
            for (int r = 0; r < 8; ++r) {
                const int m  = m0 + sub * 16 + r + 8 * g;
                const int bi = m >> 11, l = m & 2047;
                const _Float16 val = (_Float16)((acc[r] + bb) * scale);
                if (vtrans)
                    out[(((size_t)bi * NHEAD + hh) * DKH + dk) * SEQ + l] = val;
                else
                    out[(((size_t)bi * NHEAD + hh) * SEQ + l) * DKH + dk] = val;
            }
        }
    }
}

// ---------------------------------------------------------------------------
// Flash attention with TDM-staged K/V tiles in LDS.
// Block = 8 waves, all on one (b,h); wave w owns 16 query rows.
// K: [B,H,L,DK] f16 (Q pre-scaled by 1/8).  V: [B,H,DK,L] f16 (transposed).
// grid = 32 heads * 16 blocks/head = 512 blocks.
// ---------------------------------------------------------------------------
#define PST 72
__global__ __launch_bounds__(256) void attn_kernel(
    const _Float16* __restrict__ Q, const _Float16* __restrict__ K,
    const _Float16* __restrict__ V, _Float16* __restrict__ X)
{
    __shared__ _Float16 ldsK[KCHUNK * DKH];   // [key][dk]  8 KB
    __shared__ _Float16 ldsV[DKH * KCHUNK];   // [dk][key]  8 KB
    __shared__ _Float16 ldsP[8][16 * PST];    // per-wave P staging

    const int lane = threadIdx.x & 31;
    const int wave = threadIdx.x >> 5;
    const int g  = lane >> 4;
    const int ln = lane & 15;

    const int bh = blockIdx.x >> 4;                    // b*16+h
    const int qt = (blockIdx.x & 15) * 8 + wave;       // query tile 0..127
    const int b  = bh >> 4, h = bh & 15;

    const _Float16* Qh = Q + ((size_t)bh * SEQ + qt * 16) * DKH;
    const _Float16* Kh = K + (size_t)bh * SEQ * DKH;
    const _Float16* Vh = V + (size_t)bh * DKH * SEQ;   // transposed head
    _Float16* myP = ldsP[wave];
    const unsigned kOfs = lds_byte_offset(ldsK);
    const unsigned vOfs = lds_byte_offset(ldsV);

    // Q A-tiles (dk 0..31 and 32..63)
    v16h qa0, qa1;
    {
        const _Float16* row = Qh + ln * DKH;
        v8h a0 = *(const v8h*)(row + 16 * g);
        v8h a1 = *(const v8h*)(row + 8 + 16 * g);
        v8h a2 = *(const v8h*)(row + 32 + 16 * g);
        v8h a3 = *(const v8h*)(row + 40 + 16 * g);
        #pragma unroll
        for (int i = 0; i < 8; ++i) {
            qa0[i] = a0[i]; qa0[i + 8] = a1[i];
            qa1[i] = a2[i]; qa1[i + 8] = a3[i];
        }
    }

    float mrow[8], lrow[8];
    v8f acc0 = {}, acc1 = {}, acc2 = {}, acc3 = {};
    #pragma unroll
    for (int r = 0; r < 8; ++r) { mrow[r] = -1.0e30f; lrow[r] = 0.0f; }

    for (int c = 0; c < SEQ / KCHUNK; ++c) {
        const int key0 = c * KCHUNK;

        __syncthreads();                 // previous tile fully consumed
        if (wave == 0) {
            // K tile: 64 keys x 64 dk, contiguous dk -> LDS [key][dk]
            tdm_load_2d_f16(kOfs, Kh + (size_t)key0 * DKH,
                            DKH, KCHUNK, DKH, SEQ, DKH);
            // V tile: 64 dk rows x 64 keys (contiguous l) -> LDS [dk][key]
            tdm_load_2d_f16(vOfs, Vh + (size_t)key0,
                            KCHUNK, DKH, SEQ, DKH, SEQ);
            __builtin_amdgcn_s_wait_tensorcnt(0);
        }
        __syncthreads();                 // tiles visible to all waves

        // ---- scores: S[16 x 64] = Q @ K^T, 4 N-tiles x chained K=64 ----
        v8f s0 = {}, s1 = {}, s2 = {}, s3 = {};
        #pragma unroll
        for (int nt = 0; nt < 4; ++nt) {
            const _Float16* kb = ldsK + (size_t)(nt * 16 + ln) * DKH + 16 * g;
            v8f* s = (nt == 0) ? &s0 : (nt == 1) ? &s1 : (nt == 2) ? &s2 : &s3;
            *s = wmma_f16(qa0, *(const v16h*)(kb), *s);
            *s = wmma_f16(qa1, *(const v16h*)(kb + 32), *s);
        }

        // ---- online softmax; lane's row r is m = r + 8*g ----
        float corr[8];
        #pragma unroll
        for (int r = 0; r < 8; ++r) {
            float v = fmaxf(fmaxf(s0[r], s1[r]), fmaxf(s2[r], s3[r]));
            #pragma unroll
            for (int mask = 1; mask < 16; mask <<= 1) v = fmaxf(v, __shfl_xor(v, mask, 16));
            const float nm = fmaxf(mrow[r], v);
            const float e0 = __expf(s0[r] - nm);
            const float e1 = __expf(s1[r] - nm);
            const float e2 = __expf(s2[r] - nm);
            const float e3 = __expf(s3[r] - nm);
            float rs = (e0 + e1) + (e2 + e3);
            #pragma unroll
            for (int mask = 1; mask < 16; mask <<= 1) rs += __shfl_xor(rs, mask, 16);
            const float cr = __expf(mrow[r] - nm);
            lrow[r] = lrow[r] * cr + rs;
            mrow[r] = nm;
            corr[r] = cr;
            _Float16* prow = myP + (size_t)(r + 8 * g) * PST + ln;
            prow[0]  = (_Float16)e0;
            prow[16] = (_Float16)e1;
            prow[32] = (_Float16)e2;
            prow[48] = (_Float16)e3;
        }
        #pragma unroll
        for (int r = 0; r < 8; ++r) {
            acc0[r] *= corr[r]; acc1[r] *= corr[r];
            acc2[r] *= corr[r]; acc3[r] *= corr[r];
        }

        // ---- P in A layout (wave-private LDS tile; DS ops wave-ordered) ----
        v16h pa0, pa1;
        #pragma unroll
        for (int i = 0; i < 16; ++i) {
            const int kl = i + 16 * g;
            pa0[i] = myP[(size_t)ln * PST + kl];
            pa1[i] = myP[(size_t)ln * PST + 32 + kl];
        }

        // ---- O += P @ V : per dk tile, 2 chained K=32 WMMAs ----
        #pragma unroll
        for (int nt = 0; nt < 4; ++nt) {
            const _Float16* vb = ldsV + (size_t)(nt * 16 + ln) * KCHUNK + 16 * g;
            v8f* acc = (nt == 0) ? &acc0 : (nt == 1) ? &acc1 : (nt == 2) ? &acc2 : &acc3;
            *acc = wmma_f16(pa0, *(const v16h*)(vb), *acc);
            *acc = wmma_f16(pa1, *(const v16h*)(vb + 32), *acc);
        }
    }

    // ---- normalize, store X[b*L + q][h*64 + dk] (f16, GEMM-A friendly) ----
    #pragma unroll
    for (int r = 0; r < 8; ++r) {
        const float inv = 1.0f / lrow[r];
        const int q = qt * 16 + r + 8 * g;
        _Float16* xrow = X + ((size_t)b * SEQ + q) * DMODEL + h * DKH;
        xrow[0 * 16 + ln] = (_Float16)(acc0[r] * inv);
        xrow[1 * 16 + ln] = (_Float16)(acc1[r] * inv);
        xrow[2 * 16 + ln] = (_Float16)(acc2[r] * inv);
        xrow[3 * 16 + ln] = (_Float16)(acc3[r] * inv);
    }
}

// ---------------------------------------------------------------------------
// Output projection (f16 operands): out = X @ Wo^T + bo, f32 out [4096,1024].
// Wave tile 32 x 64, grid = (128, 2).
// ---------------------------------------------------------------------------
__global__ __launch_bounds__(256) void out_proj_kernel(
    const _Float16* __restrict__ X, const _Float16* __restrict__ W,
    const float* __restrict__ bias, float* __restrict__ out)
{
    const int lane = threadIdx.x & 31;
    const int wave = threadIdx.x >> 5;
    const int g  = lane >> 4;
    const int ln = lane & 15;
    const int m0 = blockIdx.x * 32;
    const int n0 = blockIdx.y * 512 + wave * 64;

    const _Float16* xrow0 = X + (size_t)(m0 + ln) * DMODEL;
    const _Float16* xrow1 = X + (size_t)(m0 + 16 + ln) * DMODEL;

    v8f aA0 = {}, aA1 = {}, aA2 = {}, aA3 = {};
    v8f aB0 = {}, aB1 = {}, aB2 = {}, aB3 = {};

    for (int kk = 0; kk < DMODEL; kk += 32) {
        __builtin_prefetch(xrow0 + kk + 256, 0, 0);
        const v16h a0 = load_a16(xrow0, kk, g);
        const v16h a1 = load_a16(xrow1, kk, g);
        #pragma unroll
        for (int nt = 0; nt < 4; ++nt) {
            const v16h bm = *(const v16h*)(W + (size_t)(n0 + nt * 16 + ln) * DMODEL + kk + 16 * g);
            v8f* pA = (nt == 0) ? &aA0 : (nt == 1) ? &aA1 : (nt == 2) ? &aA2 : &aA3;
            v8f* pB = (nt == 0) ? &aB0 : (nt == 1) ? &aB1 : (nt == 2) ? &aB2 : &aB3;
            *pA = wmma_f16(a0, bm, *pA);
            *pB = wmma_f16(a1, bm, *pB);
        }
    }

    #pragma unroll
    for (int nt = 0; nt < 4; ++nt) {
        const int n = n0 + nt * 16 + ln;
        const float bb = bias[n];
        const v8f accA = (nt == 0) ? aA0 : (nt == 1) ? aA1 : (nt == 2) ? aA2 : aA3;
        const v8f accB = (nt == 0) ? aB0 : (nt == 1) ? aB1 : (nt == 2) ? aB2 : aB3;
        #pragma unroll
        for (int r = 0; r < 8; ++r) {
            out[(size_t)(m0 + r + 8 * g) * DMODEL + n]      = accA[r] + bb;
            out[(size_t)(m0 + 16 + r + 8 * g) * DMODEL + n] = accB[r] + bb;
        }
    }
}

// ---------------------------------------------------------------------------
extern "C" void kernel_launch(void* const* d_in, const int* in_sizes, int n_in,
                              void* d_out, int out_size, void* d_ws, size_t ws_size,
                              hipStream_t stream)
{
    const float* q  = (const float*)d_in[0];
    const float* k  = (const float*)d_in[1];
    const float* v  = (const float*)d_in[2];
    const float* Wq = (const float*)d_in[3];
    const float* bq = (const float*)d_in[4];
    const float* Wk = (const float*)d_in[5];
    const float* bk = (const float*)d_in[6];
    const float* Wv = (const float*)d_in[7];
    const float* bv = (const float*)d_in[8];
    const float* Wo = (const float*)d_in[9];
    const float* bo = (const float*)d_in[10];
    float* out = (float*)d_out;

    const size_t xel = (size_t)MROWS * DMODEL;      // 4M
    const size_t wel = (size_t)DMODEL * DMODEL;     // 1M
    _Float16* Qb  = (_Float16*)d_ws;                // 4M f16 each
    _Float16* Kb  = Qb  + xel;
    _Float16* Vb  = Kb  + xel;
    _Float16* Xb  = Vb  + xel;
    _Float16* qh  = Xb  + xel;
    _Float16* kh  = qh  + xel;
    _Float16* vh  = kh  + xel;
    _Float16* Wqh = vh  + xel;                      // 1M f16 each
    _Float16* Wkh = Wqh + wel;
    _Float16* Wvh = Wkh + wel;
    _Float16* Woh = Wvh + wel;

    dim3 blk(256);
    const int xblks = (int)(xel / 8 / 256);         // 2048
    const int wblks = (int)(wel / 8 / 256);         // 512

    cvt_kernel<<<xblks, blk, 0, stream>>>(q,  qh,  (int)(xel / 8));
    cvt_kernel<<<xblks, blk, 0, stream>>>(k,  kh,  (int)(xel / 8));
    cvt_kernel<<<xblks, blk, 0, stream>>>(v,  vh,  (int)(xel / 8));
    cvt_kernel<<<wblks, blk, 0, stream>>>(Wq, Wqh, (int)(wel / 8));
    cvt_kernel<<<wblks, blk, 0, stream>>>(Wk, Wkh, (int)(wel / 8));
    cvt_kernel<<<wblks, blk, 0, stream>>>(Wv, Wvh, (int)(wel / 8));
    cvt_kernel<<<wblks, blk, 0, stream>>>(Wo, Woh, (int)(wel / 8));

    dim3 gemmGrid(MROWS / 32, DMODEL / 512);
    const float scale = 0.125f;   // 1/sqrt(64), folded into Q

    proj_kernel<<<gemmGrid, blk, 0, stream>>>(qh, Wqh, bq, Qb, scale, 0);
    proj_kernel<<<gemmGrid, blk, 0, stream>>>(kh, Wkh, bk, Kb, 1.0f, 0);
    proj_kernel<<<gemmGrid, blk, 0, stream>>>(vh, Wvh, bv, Vb, 1.0f, 1);

    attn_kernel<<<dim3(512), blk, 0, stream>>>(Qb, Kb, Vb, Xb);

    out_proj_kernel<<<gemmGrid, blk, 0, stream>>>(Xb, Woh, bo, out);
}
